// Task_Attn_15908558864909
// MI455X (gfx1250) — compile-verified, matrix operand order checked by reading, not run on hardware
//
#include <hip/hip_runtime.h>
#include <hip/hip_bf16.h>

// ---------------------------------------------------------------------------
// B=8, N=4096, HQ=16, HK=32, D=64
// Memory-bound task (~320MB traffic vs ~15 GFLOP): one-pass read of x, f16
// WMMA for all three GEMM stages, weights prepacked into fragment order.
// ---------------------------------------------------------------------------

typedef _Float16 f16;
typedef __attribute__((ext_vector_type(16))) _Float16 v16h;
typedef __attribute__((ext_vector_type(8)))  _Float16 v8h;
typedef __attribute__((ext_vector_type(4)))  _Float16 v4h;
typedef __attribute__((ext_vector_type(2)))  _Float16 h2;
typedef __attribute__((ext_vector_type(8)))  float    v8f;

#define BQ   8
#define NQ   4096
#define HQ   16
#define HK   32
#define DD   64
#define TOK  16            // tokens per workgroup tile
#define NTHREADS 256       // 8 waves

static __device__ __forceinline__ v16h cat16(v8h lo, v8h hi) {
    v16h r;
#pragma unroll
    for (int i = 0; i < 8; ++i) { r[i] = lo[i]; r[i + 8] = hi[i]; }
    return r;
}

static __device__ __forceinline__ v8f wmma_f16(v16h a, v16h b, v8f c) {
    // v_wmma_f32_16x16x32_f16  (8-arg form, probe-confirmed)
    return __builtin_amdgcn_wmma_f32_16x16x32_f16(
        false, a, false, b, (short)0, c, false, false);
}

// A-fragment (16x32 f16, M=rows of `p`, K = kc*32..kc*32+31) from LDS.
// Documented layout: lane&15 = row M; lanes 0-15 K=0-7,16-23; lanes 16-31
// K=8-15,24-31 -> two contiguous 16B reads per lane.
static __device__ __forceinline__ v16h loadA_lds(const f16* p, int rowStride,
                                                 int kc, int lane) {
    const int row = lane & 15;
    const int kb  = (lane >> 4) * 8;
    const int off = row * rowStride + kc * 32 + kb;
    v8h lo = *(const v8h*)(p + off);        // K = kc*32 + kb + 0..7
    v8h hi = *(const v8h*)(p + off + 16);   // K = kc*32 + 16 + kb + 0..7
    return cat16(lo, hi);
}

// B-fragment of prepacked weights: ((h*8 + et*2 + kc)*2 + half)*32 + lane,
// 8 f16 each -> two coalesced 16B global loads per lane.
static __device__ __forceinline__ v16h loadB_w(const f16* wpack, int h, int et,
                                               int kc, int lane) {
    const int fh   = h * 8 + et * 2 + kc;
    const f16* b0  = wpack + ((fh * 2 + 0) * 32 + lane) * 8;
    const f16* b1  = wpack + ((fh * 2 + 1) * 32 + lane) * 8;
    return cat16(*(const v8h*)b0, *(const v8h*)b1);
}

// B-fragment of K^T from LDS (K=e, N=hk). KT layout: [t][e/2][hk][2] f16.
static __device__ __forceinline__ v16h loadB_kt(const f16* kt, int t, int nt,
                                                int kc, int lane) {
    const int hk = nt * 16 + (lane & 15);
    const int kb = (lane >> 4) * 8;
    v16h r;
#pragma unroll
    for (int v = 0; v < 8; ++v) {
        const int e = kc * 32 + kb + ((v < 4) ? 2 * v : 16 + 2 * (v - 4));
        h2 x = *(const h2*)(kt + t * 2048 + (e >> 1) * 64 + hk * 2);
        r[2 * v]     = x[0];
        r[2 * v + 1] = x[1];
    }
    return r;
}

// ---------------------------------------------------------------------------
// Prep: repack Wq[16,64,64] / Wk[32,64,64] f32 -> f16 fragment-major buffers.
// One thread per output f16. 48*4096 elements total.
// ---------------------------------------------------------------------------
__global__ __launch_bounds__(256) void pack_weights_kernel(
    const float* __restrict__ Wq, const float* __restrict__ Wk,
    f16* __restrict__ pq, f16* __restrict__ pk) {
    const int idx = blockIdx.x * 256 + threadIdx.x;   // [0, 48*4096)
    const int hh  = idx >> 12;
    const int r   = idx & 4095;
    const int p    = r & 1;
    const int j    = (r >> 1) & 3;
    const int lane = (r >> 3) & 31;
    const int half = (r >> 8) & 1;
    const int fh   = r >> 9;          // et*2 + kc
    const int et   = fh >> 1;
    const int kc   = fh & 1;
    const int e = et * 16 + (lane & 15);
    const int d = kc * 32 + (half ? 16 : 0) + 2 * j + ((lane >> 4) * 8) + p;
    if (hh < HQ) {
        pq[hh * 4096 + r] = (f16)Wq[hh * 4096 + e * 64 + d];
    } else {
        const int h = hh - HQ;
        pk[h * 4096 + r] = (f16)Wk[h * 4096 + e * 64 + d];
    }
}

// ---------------------------------------------------------------------------
// Main fused kernel: 1 WG = 16 tokens.
// LDS (dynamic, 160KB): XN[16][1024] | XO[16][1024] | Q[16][16*64] | KT[16][2048]
// ---------------------------------------------------------------------------
__global__ __launch_bounds__(NTHREADS) void attn_fused_kernel(
    const float* __restrict__ xold, const float* __restrict__ xnew,
    const f16* __restrict__ wqp, const f16* __restrict__ wkp,
    const float* __restrict__ lam, float* __restrict__ out) {
    extern __shared__ f16 smem[];
    f16* XN = smem;              // 16384 f16
    f16* XO = smem + 16384;      // 16384 f16
    f16* Qs = smem + 32768;      // 16384 f16  [t][hq*64 + e]
    f16* KT = smem + 49152;      // 32768 f16  [t][e/2][hk][2]

    const int tid  = threadIdx.x;
    const int lane = tid & 31;
    const int wave = tid >> 5;
    const int g0   = blockIdx.x * TOK;      // first global token of tile
    const float lamv = lam[0];

    // ---- stage x tiles to LDS as f16 (B128 global loads, B64 LDS writes) ----
    {
        const float4* xn4 = (const float4*)(xnew + (size_t)g0 * 1024);
        const float4* xo4 = (const float4*)(xold + (size_t)g0 * 1024);
#pragma unroll
        for (int i = 0; i < 16; ++i) {
            const int idx = tid + i * NTHREADS;      // float4 units
            float4 a = xn4[idx];
            float4 b = xo4[idx];
            v4h ha, hb;
            ha[0] = (f16)a.x; ha[1] = (f16)a.y; ha[2] = (f16)a.z; ha[3] = (f16)a.w;
            hb[0] = (f16)b.x; hb[1] = (f16)b.y; hb[2] = (f16)b.z; hb[3] = (f16)b.w;
            *(v4h*)(XN + idx * 4) = ha;
            *(v4h*)(XO + idx * 4) = hb;
        }
    }
    __syncthreads();

    const int row = lane & 15;
    const int mhi = (lane >> 4) * 8;   // +8 rows for lanes 16-31 in C/D layout

    // ---- Phase Q: q = x_new @ Wq[h]^T * D^-0.5, M=16 tokens, 2 heads/wave ----
#pragma unroll
    for (int hh = 0; hh < 2; ++hh) {
        const int h = wave * 2 + hh;
#pragma unroll
        for (int et = 0; et < 4; ++et) {
            v8f acc;
#pragma unroll
            for (int i = 0; i < 8; ++i) acc[i] = 0.0f;
#pragma unroll
            for (int kc = 0; kc < 2; ++kc) {
                v16h a = loadA_lds(XN + h * 64, 1024, kc, lane);
                v16h b = loadB_w(wqp, h, et, kc, lane);
                acc = wmma_f16(a, b, acc);
            }
            const int e = et * 16 + row;
#pragma unroll
            for (int r = 0; r < 8; ++r) {
                const int t = r + mhi;
                Qs[t * 1024 + h * 64 + e] = (f16)(acc[r] * 0.125f);  // D^-0.5
            }
        }
    }
    __syncthreads();

    // ---- Phase K: k = [x_old|x_new] @ Wk[hk]^T, 4 heads/wave, store K^T ----
#pragma unroll
    for (int hh = 0; hh < 4; ++hh) {
        const int hk = wave * 4 + hh;
        const f16* src = (hk < HQ) ? XO : XN;
        const int hl = hk & 15;
#pragma unroll
        for (int et = 0; et < 4; ++et) {
            v8f acc;
#pragma unroll
            for (int i = 0; i < 8; ++i) acc[i] = 0.0f;
#pragma unroll
            for (int kc = 0; kc < 2; ++kc) {
                v16h a = loadA_lds(src + hl * 64, 1024, kc, lane);
                v16h b = loadB_w(wkp, hk, et, kc, lane);
                acc = wmma_f16(a, b, acc);
            }
            const int e = et * 16 + row;
#pragma unroll
            for (int r = 0; r < 8; ++r) {
                const int t = r + mhi;
                KT[t * 2048 + (e >> 1) * 64 + hk * 2 + (e & 1)] = (f16)acc[r];
            }
        }
    }
    __syncthreads();

    // ---- Phase S: per-token scores Q[16x64] @ K^T[64x32], softmax, x lam ----
#pragma unroll
    for (int tt = 0; tt < 2; ++tt) {
        const int t = wave * 2 + tt;
        v8f acc0, acc1;
#pragma unroll
        for (int i = 0; i < 8; ++i) { acc0[i] = 0.0f; acc1[i] = 0.0f; }
#pragma unroll
        for (int kc = 0; kc < 2; ++kc) {
            v16h a  = loadA_lds(Qs + t * 1024, 64, kc, lane);  // M=hq, K=e
            v16h b0 = loadB_kt(KT, t, 0, kc, lane);            // N=hk 0-15
            v16h b1 = loadB_kt(KT, t, 1, kc, lane);            // N=hk 16-31
            acc0 = wmma_f16(a, b0, acc0);
            acc1 = wmma_f16(a, b1, acc1);
        }
        float* obase = out + ((size_t)(g0 + t)) * (HQ * HK);
#pragma unroll
        for (int r = 0; r < 8; ++r) {
            const int m = r + mhi;                    // q-head (row)
            // row m lives entirely in this 16-lane half: cols 0-15 in acc0[r],
            // cols 16-31 in acc1[r]. Reduce across lanes with xor 1,2,4,8.
            float mx = fmaxf(acc0[r], acc1[r]);
#pragma unroll
            for (int s = 1; s < 16; s <<= 1)
                mx = fmaxf(mx, __shfl_xor(mx, s, 32));
            const float e0 = __expf(acc0[r] - mx);
            const float e1 = __expf(acc1[r] - mx);
            float sum = e0 + e1;
#pragma unroll
            for (int s = 1; s < 16; s <<= 1)
                sum += __shfl_xor(sum, s, 32);
            const float inv = lamv / sum;
            obase[m * HK + row]      = e0 * inv;
            obase[m * HK + 16 + row] = e1 * inv;
        }
    }
}

extern "C" void kernel_launch(void* const* d_in, const int* in_sizes, int n_in,
                              void* d_out, int out_size, void* d_ws, size_t ws_size,
                              hipStream_t stream) {
    const float* xold = (const float*)d_in[0];
    const float* xnew = (const float*)d_in[1];
    const float* Wq   = (const float*)d_in[2];
    const float* Wk   = (const float*)d_in[3];
    const float* lam  = (const float*)d_in[4];
    float* out = (float*)d_out;

    // workspace: packed f16 weights (Wq: 128KB, Wk: 256KB)
    f16* pq = (f16*)d_ws;
    f16* pk = pq + HQ * 4096;

    // 1) repack weights into WMMA B-fragment order (L2-resident thereafter)
    pack_weights_kernel<<<(48 * 4096) / 256, 256, 0, stream>>>(Wq, Wk, pq, pk);

    // 2) fused projections + scores + softmax
    const int nblocks = (BQ * NQ) / TOK;          // 2048
    const size_t ldsBytes = 81920 * sizeof(f16);  // 160 KB dynamic LDS
    attn_fused_kernel<<<nblocks, NTHREADS, ldsBytes, stream>>>(
        xold, xnew, pq, pk, lam, out);
}
